// RNNQNetwork_7885559955595
// MI455X (gfx1250) — compile-verified, matrix operand order checked by reading, not run on hardware
//
#include <hip/hip_runtime.h>
#include <hip/hip_bf16.h>

typedef _Float16 half_t;
typedef __attribute__((ext_vector_type(16))) _Float16 v16h;
typedef __attribute__((ext_vector_type(8)))  _Float16 v8h;
typedef __attribute__((ext_vector_type(8)))  float    v8f;
typedef __attribute__((ext_vector_type(4)))  unsigned int u32x4;
typedef __attribute__((ext_vector_type(8)))  int          i32x8;
typedef __attribute__((ext_vector_type(4)))  int          i32x4;

#define B_SZ 64
#define T_SZ 2048
#define I_SZ 128
#define H_SZ 256
#define A_SZ 18
#define BT   (B_SZ * T_SZ)   /* 131072 rows */
#define WLD  264             /* padded LDS stride for W_hh (halves): 512B row + 16B pad */
#define HPD  264             /* padded LDS stride for h buffers (halves) */

#if defined(__has_builtin)
#if __has_builtin(__builtin_amdgcn_tensor_load_to_lds) && \
    __has_builtin(__builtin_amdgcn_s_wait_tensorcnt)
#define HAVE_TDM 1
#endif
#endif

// ---- WMMA fragment helpers (layouts per CDNA5 ISA 7.12.2, wave32) ----
// A-frag / B-frag: lane L (<16) holds K = {kh..kh+7} U {kh+16..kh+23},
// kh = 8*(L>=16).  Caller passes p = &src[row*ld + kbase + kh].
static __device__ __forceinline__ v16h ld_frag_f16(const half_t* p) {
  v16h r;
#pragma unroll
  for (int i = 0; i < 8; ++i) { r[i] = p[i]; r[i + 8] = p[i + 16]; }
  return r;
}
static __device__ __forceinline__ v16h ld_frag_f32(const float* p) {
  v16h r;
#pragma unroll
  for (int i = 0; i < 8; ++i) { r[i] = (half_t)p[i]; r[i + 8] = (half_t)p[i + 16]; }
  return r;
}
static __device__ __forceinline__ v8f wmma16(v16h a, v16h b, v8f c) {
  return __builtin_amdgcn_wmma_f32_16x16x32_f16(false, a, false, b, (short)0, c,
                                                false, false);
}

// ---- Kernel 0: one-time fp32 -> fp16 weight conversion (+ pad W_out to 32 rows)
__global__ void __launch_bounds__(256)
cvt_weights(const float* Wi, const float* Wih, const float* Whh, const float* Wo,
            half_t* wi_h, half_t* wih_h, half_t* whh_h, half_t* wo_h) {
  int idx = blockIdx.x * blockDim.x + threadIdx.x;
  int stride = gridDim.x * blockDim.x;
  const int n1 = H_SZ * I_SZ, n2 = H_SZ * H_SZ;
  for (int i = idx; i < n1; i += stride) wi_h[i] = (half_t)Wi[i];
  for (int i = idx; i < n2; i += stride) wih_h[i] = (half_t)Wih[i];
  for (int i = idx; i < n2; i += stride) whh_h[i] = (half_t)Whh[i];
  for (int i = idx; i < 32 * H_SZ; i += stride) {
    int r = i / H_SZ;
    wo_h[i] = (r < A_SZ) ? (half_t)Wo[i] : (half_t)0.0f;
  }
}

// ---- Kernel A: h_in = relu(x @ W_init^T + b_init), fp16 out, rows = b*T+t
__global__ void __launch_bounds__(512)
gemm_init(const float* __restrict__ x, const half_t* __restrict__ wi_h,
          const float* __restrict__ b_init, half_t* __restrict__ h_in) {
  const int lane  = threadIdx.x & 31;
  const int wave  = threadIdx.x >> 5;
  const int mbase = blockIdx.x * 16;
  const int nbase = wave * 16;
  const int kh    = (lane >> 4) * 8;
  const int arow  = mbase + (lane & 15);
  const int wrow  = nbase + (lane & 15);
  v8f c = {};
#pragma unroll
  for (int k = 0; k < I_SZ; k += 32) {
    v16h a = ld_frag_f32(x    + (size_t)arow * I_SZ + k + kh);
    v16h b = ld_frag_f16(wi_h + (size_t)wrow * I_SZ + k + kh);
    c = wmma16(a, b, c);
  }
  const int hi = lane >> 4;
  const int n  = nbase + (lane & 15);
  const float bias = b_init[n];
#pragma unroll
  for (int r = 0; r < 8; ++r) {
    int m = r + 8 * hi;
    float v = c[r] + bias;
    v = v > 0.f ? v : 0.f;
    h_in[(size_t)(mbase + m) * H_SZ + n] = (half_t)v;
  }
}

// ---- Kernel B: xW_T = transpose_to_[T,B,H]( h_in @ W_ih^T + (b_ih + b_hh) )
// Input rows are b*T+t; output row is t*B+b so the scan reads contiguous tiles.
__global__ void __launch_bounds__(512)
gemm_ih(const half_t* __restrict__ h_in, const half_t* __restrict__ wih_h,
        const float* __restrict__ b_ih, const float* __restrict__ b_hh,
        float* __restrict__ xWT) {
  const int lane  = threadIdx.x & 31;
  const int wave  = threadIdx.x >> 5;
  const int mbase = blockIdx.x * 16;
  const int nbase = wave * 16;
  const int kh    = (lane >> 4) * 8;
  const int arow  = mbase + (lane & 15);
  const int wrow  = nbase + (lane & 15);
  v8f c = {};
#pragma unroll
  for (int k = 0; k < H_SZ; k += 32) {
    v16h a = ld_frag_f16(h_in  + (size_t)arow * H_SZ + k + kh);
    v16h b = ld_frag_f16(wih_h + (size_t)wrow * H_SZ + k + kh);
    c = wmma16(a, b, c);
  }
  const int hi = lane >> 4;
  const int n  = nbase + (lane & 15);
  const float bias = b_ih[n] + b_hh[n];
#pragma unroll
  for (int r = 0; r < 8; ++r) {
    int row = mbase + r + 8 * hi;     // row = b*T + t
    int b = row >> 11;                // /T (2048)
    int t = row & (T_SZ - 1);
    xWT[((size_t)t * B_SZ + b) * H_SZ + n] = c[r] + bias;
  }
}

// ---- Kernel C: sequential scan.  4 blocks (one 16-row batch tile each),
// 512 threads = 16 waves (one N-tile each).  W_hh staged into LDS by the
// Tensor Data Mover (with HW padding to a conflict-free stride) when the
// builtin is available; h double-buffered in LDS; xW/hs in [T,B,H] layout
// so each step touches one contiguous block.
__global__ void __launch_bounds__(512)
rnn_scan(const float* __restrict__ xWT, const half_t* __restrict__ whh_h,
         half_t* __restrict__ hsT) {
  extern __shared__ char smem[];
  half_t* w_s = (half_t*)smem;                                          // [H][WLD]
  half_t* hb  = (half_t*)(smem + (size_t)H_SZ * WLD * sizeof(half_t));  // [2][16][HPD]

  const int tid   = threadIdx.x;
  const int lane  = tid & 31;
  const int wave  = tid >> 5;
  const int bbase = blockIdx.x * 16;

#if defined(HAVE_TDM)
  // TDM: DMA W_hh (256 rows x 512B) global->LDS, padding 16B after every
  // 512B row => LDS stride 264 halves (bank-conflict-free for B-frag reads).
  if (wave == 0) {
    unsigned long long ga = (unsigned long long)(uintptr_t)whh_h;
    unsigned int lds_off  = (unsigned int)(uintptr_t)w_s;  // low 32b = LDS offset
    u32x4 g0;
    g0[0] = 1u;                                        // count=1 (valid user D#)
    g0[1] = lds_off;                                   // lds_addr (bytes)
    g0[2] = (unsigned int)(ga & 0xFFFFFFFFull);        // global_addr[31:0]
    g0[3] = (unsigned int)((ga >> 32) & 0x1FFFFFFull)  // global_addr[56:32]
          | (2u << 30);                                // type=2 ("image")
    i32x8 g1 = {};
    // mask=0 | data_size=3 (8B) | pad_enable | pad_interval=6 (128 DW = 512B)
    // | pad_amount=3 (4 DW = 16B)
    g1[0] = (3 << 16) | (1 << 20) | (6 << 22) | (3 << 25);
    g1[1] = (64 << 16);    // tensor_dim0 = 64 (8B units) -> bits[63:48]
    g1[2] = (256 << 16);   // tensor_dim0 hi=0; tensor_dim1 = 256 -> bits[95:80]
    g1[3] = (64 << 16);    // tensor_dim1 hi=0; tile_dim0 = 64 -> bits[127:112]
    g1[4] = 256;           // tile_dim1 = 256 -> bits[143:128]; tile_dim2 = 0
    g1[5] = 64;            // tensor_dim0_stride = 64 (8B units)
    i32x4 gz = {};
#if __clang_major__ >= 23
    i32x8 gz8 = {};
    __builtin_amdgcn_tensor_load_to_lds(g0, g1, gz, gz, gz8, 0);
#else
    __builtin_amdgcn_tensor_load_to_lds(g0, g1, gz, gz, 0);
#endif
    __builtin_amdgcn_s_wait_tensorcnt(0);
  }
#else
  for (int g = tid; g < (H_SZ * H_SZ) / 8; g += 512) {
    int row = g >> 5;
    int col = (g & 31) * 8;
    *(v8h*)(w_s + row * WLD + col) = *(const v8h*)(whh_h + row * H_SZ + col);
  }
#endif
  // h0 = 0 in buffer 0
  for (int i = tid; i < 16 * HPD; i += 512) hb[i] = (half_t)0.0f;
  __syncthreads();

  const int nbase = wave * 16;
  const int kh    = (lane >> 4) * 8;
  const int hi    = lane >> 4;
  const int n     = nbase + (lane & 15);

  auto step = [&](const half_t* hcur, half_t* hnxt, int t) {
    // C operand straight from contiguous xW_T tile (biases pre-folded)
    const float* ctile = xWT + ((size_t)t * B_SZ + bbase) * H_SZ;
    v8f c;
#pragma unroll
    for (int r = 0; r < 8; ++r) {
      int m = r + 8 * hi;
      c[r] = ctile[(size_t)m * H_SZ + n];
    }
    if (t + 1 < T_SZ) {   // one prefetch/thread covers the next 16KB tile
      const float* nt = xWT + ((size_t)(t + 1) * B_SZ + bbase) * H_SZ;
      __builtin_prefetch(nt + (size_t)tid * 8, 0, 3);
    }
#pragma unroll
    for (int k = 0; k < H_SZ; k += 32) {
      v16h a = ld_frag_f16(hcur + (lane & 15) * HPD + k + kh);
      v16h b = ld_frag_f16(w_s + n * WLD + k + kh);
      c = wmma16(a, b, c);
    }
    half_t* hrow = hsT + ((size_t)t * B_SZ + bbase) * H_SZ;
#pragma unroll
    for (int r = 0; r < 8; ++r) {
      int m = r + 8 * hi;
      float v = c[r] > 0.f ? c[r] : 0.f;
      half_t hv = (half_t)v;
      hnxt[m * HPD + n] = hv;                 // feed recurrence
      hrow[(size_t)m * H_SZ + n] = hv;        // feed output GEMM (contiguous)
    }
    __syncthreads();
  };

  half_t* h0 = hb;
  half_t* h1 = hb + 16 * HPD;
  for (int t = 0; t < T_SZ; t += 2) {   // unroll-by-2: static buffer parity
    step(h0, h1, t);
    step(h1, h0, t + 1);
  }
}

// ---- Kernel D: out = hs @ W_out^T + b_out.  hs rows are t*B+b; output is
// [B,T,A].  A=18 padded to 32 cols, stores masked.
__global__ void __launch_bounds__(512)
gemm_out(const half_t* __restrict__ hsT, const half_t* __restrict__ wo_h,
         const float* __restrict__ b_out, float* __restrict__ out) {
  const int lane  = threadIdx.x & 31;
  const int wave  = threadIdx.x >> 5;          // 16 waves: 8 M-tiles x 2 N-tiles
  const int mbase = (blockIdx.x * 8 + (wave >> 1)) * 16;
  const int nbase = (wave & 1) * 16;
  const int kh    = (lane >> 4) * 8;
  const int arow  = mbase + (lane & 15);
  const int wrow  = nbase + (lane & 15);
  v8f c = {};
#pragma unroll
  for (int k = 0; k < H_SZ; k += 32) {
    v16h a = ld_frag_f16(hsT  + (size_t)arow * H_SZ + k + kh);
    v16h b = ld_frag_f16(wo_h + (size_t)wrow * H_SZ + k + kh);
    c = wmma16(a, b, c);
  }
  const int hi = lane >> 4;
  const int n  = nbase + (lane & 15);
  const float bias = (n < A_SZ) ? b_out[n] : 0.f;
#pragma unroll
  for (int r = 0; r < 8; ++r) {
    int row = mbase + r + 8 * hi;   // row = t*B + b
    int t = row >> 6;               // /B (64)
    int b = row & (B_SZ - 1);
    if (n < A_SZ)
      out[((size_t)b * T_SZ + t) * A_SZ + n] = c[r] + bias;
  }
}

extern "C" void kernel_launch(void* const* d_in, const int* in_sizes, int n_in,
                              void* d_out, int out_size, void* d_ws, size_t ws_size,
                              hipStream_t stream) {
  const float* x      = (const float*)d_in[0];
  const float* W_init = (const float*)d_in[1];
  const float* b_init = (const float*)d_in[2];
  const float* W_ih   = (const float*)d_in[3];
  const float* b_ih   = (const float*)d_in[4];
  const float* W_hh   = (const float*)d_in[5];
  const float* b_hh   = (const float*)d_in[6];
  const float* W_out  = (const float*)d_in[7];
  const float* b_out  = (const float*)d_in[8];
  float* out = (float*)d_out;

  char* ws = (char*)d_ws;
  size_t off = 0;
  half_t* wi_h  = (half_t*)(ws + off); off += (size_t)H_SZ * I_SZ * 2;
  half_t* wih_h = (half_t*)(ws + off); off += (size_t)H_SZ * H_SZ * 2;
  half_t* whh_h = (half_t*)(ws + off); off += (size_t)H_SZ * H_SZ * 2;
  half_t* wo_h  = (half_t*)(ws + off); off += (size_t)32 * H_SZ * 2;
  off = (off + 255) & ~(size_t)255;
  half_t* h_in  = (half_t*)(ws + off); off += (size_t)BT * H_SZ * 2;  // reused as hsT
  float*  xWT   = (float*)(ws + off);  off += (size_t)BT * H_SZ * 4;
  half_t* hsT   = h_in;   // h_in dead after gemm_ih; reuse region

  cvt_weights<<<64, 256, 0, stream>>>(W_init, W_ih, W_hh, W_out,
                                      wi_h, wih_h, whh_h, wo_h);
  gemm_init<<<BT / 16, 512, 0, stream>>>(x, wi_h, b_init, h_in);
  gemm_ih<<<BT / 16, 512, 0, stream>>>(h_in, wih_h, b_ih, b_hh, xWT);

  const size_t smem = (size_t)H_SZ * WLD * sizeof(half_t)
                    + (size_t)2 * 16 * HPD * sizeof(half_t);
  rnn_scan<<<B_SZ / 16, 512, smem, stream>>>(xWT, whh_h, hsT);

  gemm_out<<<BT / 128, 512, 0, stream>>>(hsT, wo_h, b_out, out);
}
